// LocalAtten_28492813042213
// MI455X (gfx1250) — matrix operand order repaired and checked by reading, not compiled, not run on hardware
//
#include <hip/hip_runtime.h>
#include <cstdint>
#include <cstddef>

#define NI    128
#define ND    32
#define NH    4
#define NNEI  120
#define ATTNW_SHIFT 20.0f
#define PITCH 136   // f16 elements per LDS row (128 + 8 pad) -> 272B, kills bank conflicts

typedef __attribute__((ext_vector_type(16))) _Float16 v16h;
typedef __attribute__((ext_vector_type(8)))  _Float16 v8h;
typedef __attribute__((ext_vector_type(4)))  _Float16 v4h;
typedef __attribute__((ext_vector_type(8)))  float    v8f;

static __device__ __forceinline__ v16h join16(v8h lo, v8h hi) {
  return __builtin_shufflevector(lo, hi, 0,1,2,3,4,5,6,7,8,9,10,11,12,13,14,15);
}
static __device__ __forceinline__ v8f wmma16(v16h a, v16h b, v8f c) {
  // D = A(16x32 f16) * B(32x16 f16) + C(16x16 f32)
  return __builtin_amdgcn_wmma_f32_16x16x32_f16(false, a, false, b, (short)0, c,
                                                false, false);
}

// Async-copy one 120x128 f32 tile (61440 B) global -> LDS staging.
// Per lane: ASYNCcnt-tracked global_load_async_to_lds_b128 (GVS mode: saddr base
// + per-lane 32-bit byte offset). 3840 b128 transfers / 256 threads = 15 each.
static __device__ __forceinline__ void async_load_tile(const float* __restrict__ base,
                                                       float* stage, int tid) {
  const unsigned long long sbase = (unsigned long long)(size_t)base;
  #pragma unroll
  for (int it = 0; it < 15; ++it) {
    const int e16 = tid + it * 256;
    // generic pointer to LDS: low 32 bits are the LDS byte address
    const unsigned lds = (unsigned)(size_t)(stage + e16 * 4);
    const unsigned off = (unsigned)(e16 * 16);
    asm volatile("global_load_async_to_lds_b128 %0, %1, %2 offset:0"
                 :: "v"(lds), "v"(off), "s"(sbase)
                 : "memory");
  }
}
static __device__ __forceinline__ void wait_async0() {
  asm volatile("s_wait_asynccnt 0x0" ::: "memory");
}

// ---------------------------------------------------------------------------
// K0a: W2T[h][o][j] = sum_i Wkv[j][(32+i)*4+h] * Wh[h*128+i][o]   (fp16)
__global__ void prep_w2(const float* __restrict__ Wkv, const float* __restrict__ Wh,
                        _Float16* __restrict__ W2T) {
  int gid = blockIdx.x * 256 + threadIdx.x;            // 0..65535
  int h = gid >> 14, rem = gid & 16383;
  int o = rem >> 7,  j = rem & 127;
  float acc = 0.f;
  for (int i = 0; i < NI; ++i)
    acc += Wkv[j * 640 + 128 + i * 4 + h] * Wh[(h * NI + i) * NI + o];
  W2T[gid] = (_Float16)acc;                            // layout = [h][o][j]
}

// K0b: WkT[h][d][j] = Wkv[j][d*4+h] / sqrt(ND)   (fp16)  -- 16384 outputs
__global__ void prep_wk(const float* __restrict__ Wkv, _Float16* __restrict__ WkT) {
  int gid = blockIdx.x * 256 + threadIdx.x;            // 0..16383
  int h = gid >> 12, d = (gid >> 7) & 31, j = gid & 127;
  WkT[gid] = (_Float16)(Wkv[j * 640 + d * 4 + h] * 0.17677669529663687f);
}

// K1: per atom l: q = g1_l @ Wq ; p[h][j] = sum_d q[d*4+h]*WkT[h][d][j]  (fp16)
__global__ void qp_kernel(const float* __restrict__ g1, const float* __restrict__ Wq,
                          const _Float16* __restrict__ WkT, _Float16* __restrict__ pAll) {
  __shared__ float g1s[NI];
  __shared__ float qs[NI];
  const int l = blockIdx.x, t = threadIdx.x;           // 128 threads
  g1s[t] = g1[l * NI + t];
  __syncthreads();
  float acc = 0.f;
  for (int j = 0; j < NI; ++j) acc += g1s[j] * Wq[j * NI + t];  // coalesced over t
  qs[t] = acc;
  __syncthreads();
  for (int h = 0; h < NH; ++h) {
    float p = 0.f;
    for (int d = 0; d < ND; ++d)
      p += qs[d * 4 + h] * (float)WkT[(h * ND + d) * NI + t];
    pAll[(l * NH + h) * NI + t] = (_Float16)p;
  }
}

// ---------------------------------------------------------------------------
// K2 (main, memory-bound): one WG = 16 atoms, software-pipelined:
//   async_load(atom a+1 -> LDS staging)  ||  WMMA/softmax phases for atom a
// Per atom:
//   logits = gg1_l @ p    (WMMA f16, N=4 padded to 16)
//   softmax((x+20)*sw-20) * sw
//   t_h    = a_h @ gg1_l  (WMMA f16, heads as M)
// Then per block: out = sum_h T_h(16x128) @ W2_h + bh   (WMMA f16)
__global__ void __launch_bounds__(256) attn_main(
    const float* __restrict__ gg1, const float* __restrict__ sw,
    const _Float16* __restrict__ pAll, const _Float16* __restrict__ W2T,
    const float* __restrict__ bh, float* __restrict__ out) {
  __shared__ __align__(16) float    sStage[NNEI * NI];    // raw f32 tile (async dest)
  __shared__ __align__(16) _Float16 sG [128 * PITCH];     // gg1_l row-major [n][i]
  __shared__ __align__(16) _Float16 sGT[128 * PITCH];     // gg1_l col-major [i][n]
  __shared__ __align__(16) _Float16 sP [NH * NI];         // p  [h][j]
  __shared__ __align__(16) float    sLog[128 * 4];        // logits [n][h]
  __shared__ __align__(16) _Float16 sAttn[16 * PITCH];    // a  [h(pad16)][n(pad)]
  __shared__ __align__(16) _Float16 sT  [NH * 16 * PITCH];// t  [h][atom][j]
  __shared__ float sSw[128];

  const int tid  = threadIdx.x;
  const int lane = tid & 31;
  const int w    = tid >> 5;        // wave id 0..7
  const int mrow = lane & 15;
  const int half = lane >> 4;

  // one-time zeroing: attn pad rows/cols, gg1 pad rows 120..127
  for (int k = tid; k < 16 * PITCH; k += 256) sAttn[k] = (_Float16)0.f;
  for (int k = tid; k < 8 * PITCH;  k += 256) sG[120 * PITCH + k] = (_Float16)0.f;

  const int atomBase = blockIdx.x * 16;
  // prologue: start async copy of atom 0's tile
  async_load_tile(gg1 + (size_t)atomBase * NNEI * NI, sStage, tid);

  for (int a = 0; a < 16; ++a) {
    const int l = atomBase + a;

    // small per-atom inputs (synchronous)
    if (tid < NNEI) sSw[tid] = sw[l * NNEI + tid];
    for (int k = tid; k < NH * NI; k += 256) sP[k] = pAll[l * NH * NI + k];

    // drain this wave's async transfers, then rendezvous all waves
    wait_async0();
    __syncthreads();

    // ---- Phase C: convert staged f32 tile -> f16 in two layouts ----
    const float4* st4 = (const float4*)sStage;
    #pragma unroll
    for (int it = 0; it < 15; ++it) {                 // 15360 floats / (256*4)
      int e4 = tid + it * 256;
      float4 v = st4[e4];
      int n = e4 >> 5;
      int i = (e4 & 31) << 2;
      v4h hv = {(_Float16)v.x, (_Float16)v.y, (_Float16)v.z, (_Float16)v.w};
      *(v4h*)&sG[n * PITCH + i] = hv;
      sGT[(i + 0) * PITCH + n] = (_Float16)v.x;
      sGT[(i + 1) * PITCH + n] = (_Float16)v.y;
      sGT[(i + 2) * PITCH + n] = (_Float16)v.z;
      sGT[(i + 3) * PITCH + n] = (_Float16)v.w;
    }
    __syncthreads();

    // staging is free again: overlap next atom's HBM traffic with compute
    if (a < 15)
      async_load_tile(gg1 + (size_t)(l + 1) * NNEI * NI, sStage, tid);

    // ---- Phase Q: logits(120x4) = gg1_l(120x128) @ p(128x16pad), wave w = M-tile ----
    {
      const int mt = w;
      v8f acc = {0.f, 0.f, 0.f, 0.f, 0.f, 0.f, 0.f, 0.f};
      #pragma unroll
      for (int kt = 0; kt < 4; ++kt) {
        const int kb = kt * 32;
        const _Float16* ar = &sG[(mt * 16 + mrow) * PITCH];
        v8h alo = *(const v8h*)&ar[kb + half * 8];
        v8h ahi = *(const v8h*)&ar[kb + 16 + half * 8];
        const int hs = mrow & 3;                          // duplicate heads in cols 4..15
        v8h blo = *(const v8h*)&sP[hs * NI + kb + half * 8];
        v8h bhi = *(const v8h*)&sP[hs * NI + kb + 16 + half * 8];
        acc = wmma16(join16(alo, ahi), join16(blo, bhi), acc);
      }
      if (mrow < 4) {                                     // keep only the 4 real heads
        #pragma unroll
        for (int r = 0; r < 8; ++r)
          sLog[(mt * 16 + r + 8 * half) * 4 + mrow] = acc[r];
      }
    }
    __syncthreads();

    // ---- Phase S: softmax over 120 neighbors; wave h handles head h ----
    if (w < NH) {
      const int h = w;
      float x[4], e[4];
      float mx = -3.0e38f;
      #pragma unroll
      for (int r = 0; r < 4; ++r) {
        int n = lane + 32 * r;
        if (n < NNEI) {
          float xx = (sLog[n * 4 + h] + ATTNW_SHIFT) * sSw[n] - ATTNW_SHIFT;
          x[r] = xx; mx = fmaxf(mx, xx);
        } else x[r] = -3.0e38f;
      }
      #pragma unroll
      for (int off = 16; off > 0; off >>= 1) mx = fmaxf(mx, __shfl_xor(mx, off));
      float s = 0.f;
      #pragma unroll
      for (int r = 0; r < 4; ++r) {
        int n = lane + 32 * r;
        e[r] = (n < NNEI) ? __expf(x[r] - mx) : 0.f;
        s += e[r];
      }
      #pragma unroll
      for (int off = 16; off > 0; off >>= 1) s += __shfl_xor(s, off);
      const float inv = 1.0f / s;
      #pragma unroll
      for (int r = 0; r < 4; ++r) {
        int n = lane + 32 * r;
        if (n < NNEI) sAttn[h * PITCH + n] = (_Float16)(e[r] * inv * sSw[n]);
      }
    }
    __syncthreads();

    // ---- Phase T: t(4x128) = a(16pad x 128pad) @ gg1_l ; wave w = N(i)-tile ----
    {
      const int nt = w;
      v8f acc = {0.f, 0.f, 0.f, 0.f, 0.f, 0.f, 0.f, 0.f};
      #pragma unroll
      for (int kt = 0; kt < 4; ++kt) {
        const int kb = kt * 32;
        v8h alo = *(const v8h*)&sAttn[mrow * PITCH + kb + half * 8];
        v8h ahi = *(const v8h*)&sAttn[mrow * PITCH + kb + 16 + half * 8];
        const int ic = nt * 16 + mrow;
        v8h blo = *(const v8h*)&sGT[ic * PITCH + kb + half * 8];
        v8h bhi = *(const v8h*)&sGT[ic * PITCH + kb + 16 + half * 8];
        acc = wmma16(join16(alo, ahi), join16(blo, bhi), acc);
      }
      if (half == 0) {                                    // rows 0..3 = heads
        const int ic = nt * 16 + mrow;
        #pragma unroll
        for (int h = 0; h < NH; ++h)
          sT[(h * 16 + a) * PITCH + ic] = (_Float16)acc[h];
      }
    }
    __syncthreads();
  }

  // ---- Phase O: out(16 atoms x 128) = sum_h T_h @ W2_h + bh ; wave w = N(o)-tile ----
  {
    const int nt = w;
    v8f acc = {0.f, 0.f, 0.f, 0.f, 0.f, 0.f, 0.f, 0.f};
    #pragma unroll
    for (int h = 0; h < NH; ++h) {
      #pragma unroll
      for (int kt = 0; kt < 4; ++kt) {
        const int kb = kt * 32;
        v8h alo = *(const v8h*)&sT[(h * 16 + mrow) * PITCH + kb + half * 8];
        v8h ahi = *(const v8h*)&sT[(h * 16 + mrow) * PITCH + kb + 16 + half * 8];
        const int oc = nt * 16 + mrow;
        const _Float16* bp = W2T + ((size_t)(h * NI + oc) * NI + kb + half * 8);
        v8h blo = *(const v8h*)bp;
        v8h bhi = *(const v8h*)(bp + 16);
        acc = wmma16(join16(alo, ahi), join16(blo, bhi), acc);
      }
    }
    const int oc = nt * 16 + mrow;
    const float bias = bh[oc];
    #pragma unroll
    for (int r = 0; r < 8; ++r)
      out[(size_t)(atomBase + r + 8 * half) * NI + oc] = acc[r] + bias;
  }
}

// ---------------------------------------------------------------------------
extern "C" void kernel_launch(void* const* d_in, const int* in_sizes, int n_in,
                              void* d_out, int out_size, void* d_ws, size_t ws_size,
                              hipStream_t stream) {
  const float* g1  = (const float*)d_in[0];
  const float* gg1 = (const float*)d_in[1];
  // d_in[2] = nlist_mask: redundant — sw is pre-masked (sw = U*mask), and
  // where(mask, attnw, 0)*sw == attnw*sw since sw==0 exactly where mask==0.
  const float* sw  = (const float*)d_in[3];
  const float* Wq  = (const float*)d_in[4];
  const float* Wkv = (const float*)d_in[5];
  const float* Wh  = (const float*)d_in[6];
  const float* bh  = (const float*)d_in[7];
  float* out = (float*)d_out;

  char* ws = (char*)d_ws;
  _Float16* W2T  = (_Float16*)ws;                    // 4*128*128*2 = 131072 B
  _Float16* WkT  = (_Float16*)(ws + 131072);         // 4*32*128*2  =  32768 B
  _Float16* pAll = (_Float16*)(ws + 131072 + 32768); // 4096*512*2  = 4 MB

  const int nloc = in_sizes[3] / NNEI;               // 4096

  prep_w2<<<256, 256, 0, stream>>>(Wkv, Wh, W2T);
  prep_wk<<<64, 256, 0, stream>>>(Wkv, WkT);
  qp_kernel<<<nloc, 128, 0, stream>>>(g1, Wq, WkT, pAll);
  attn_main<<<nloc / 16, 256, 0, stream>>>(gg1, sw, pAll, W2T, bh, out);
}
